// WindowAttention_75874892251734
// MI455X (gfx1250) — matrix operand order repaired and checked by reading, not compiled, not run on hardware
//
#include <hip/hip_runtime.h>

// ---------------------------------------------------------------------------
// Window attention for [2,256,256,256] inputs, WS=8, HEADS=8, head_dim=32.
// One workgroup (8 wave32) per 64-token window; f16 WMMA everywhere.
// Input window staged via Tensor Data Mover (tensor_load_to_lds, TENSORcnt).
// ---------------------------------------------------------------------------

typedef _Float16 v16h __attribute__((ext_vector_type(16)));
typedef _Float16 v8h  __attribute__((ext_vector_type(8)));
typedef float    v8f  __attribute__((ext_vector_type(8)));
typedef int      v4i  __attribute__((ext_vector_type(4)));
typedef int      v8i  __attribute__((ext_vector_type(8)));

union F16Frag { v16h v; v8h h[2]; };

#define DIM   256
#define NTOK  64
#define T16S  264   // padded f16 row stride (528B = 33*16B: keeps b128 align, kills bank conflicts)

// f16 workspace layout (element offsets) for transposed weights: Wt[n][k]
#define OFF_WQT 0
#define OFF_WKT 65536
#define OFF_WVT 131072
#define OFF_WMT 196608
#define OFF_W1T 262144   // [512][512]
#define OFF_W2T 524288   // [256][512]

// LDS layout (byte offsets), peak 292 KB
#define LDS_TC32 0       // TDM tile, channel-major fp32 [256][64]   64KB (persistent)
#define LDS_T16  65536   // t f16 token-major [64][T16S]             36KB (persistent)
#define LDS_Q    102400  // q f16  [64][256]                         32KB
#define LDS_K    135168  // k f16  [64][256]                         32KB
#define LDS_VT   167936  // v^T f16 [256][64]                        32KB
#define LDS_ATT  200704  // attn f16 [8 waves][64][64]               64KB
#define LDS_O    266240  // attn-out f16 [64][256]                   32KB
// post-attention reuse:
#define LDS_S32  102400  // gemm f32 out [64][256]                   64KB
#define LDS_LN1  167936  // ln1 f16 [64][T16S]                       36KB
#define LDS_H1   204800  // relu f16 [64][512]                       64KB (overlaps dead ATT/O tail)
#define SMEM_BYTES 299008

__device__ __forceinline__ v8f wmma16(v16h a, v16h b, v8f c) {
  return __builtin_amdgcn_wmma_f32_16x16x32_f16(false, a, false, b,
                                                (short)0, c, false, false);
}

// A fragment (16x32 f16, row-major source, 16B-aligned chunks).
__device__ __forceinline__ v16h load_a_frag(const _Float16* base, int stride,
                                            int m0, int k0, int lane) {
  int r  = m0 + (lane & 15);
  int kk = k0 + ((lane >> 4) << 3);
  const _Float16* p = base + r * stride + kk;
  F16Frag f;
  f.h[0] = *(const v8h*)(p);
  f.h[1] = *(const v8h*)(p + 16);
  return f.v;
}

// B fragment (32x16 f16) from row-major transposed matrix Bt[n][k].
__device__ __forceinline__ v16h load_b_frag(const _Float16* base, int stride,
                                            int n0, int k0, int lane) {
  int c  = n0 + (lane & 15);
  int kk = k0 + ((lane >> 4) << 3);
  const _Float16* p = base + c * stride + kk;
  F16Frag f;
  f.h[0] = *(const v8h*)(p);
  f.h[1] = *(const v8h*)(p + 16);
  return f.v;
}

// ---------------------------------------------------------------------------
// Prep: fp32 W[k][n] -> f16 Wt[n][k]
// ---------------------------------------------------------------------------
__global__ void cvt_transpose_kernel(const float* __restrict__ src,
                                     _Float16* __restrict__ dst, int K, int N) {
  int idx = blockIdx.x * blockDim.x + threadIdx.x;
  if (idx < K * N) {
    int k = idx / N, n = idx % N;
    dst[n * K + k] = (_Float16)src[idx];
  }
}

__global__ void scalar_tail_kernel(float* __restrict__ out, int out_size) {
  if (threadIdx.x == 0 && blockIdx.x == 0) {
    const long long base = 2LL * 2048 * 64 * 256;
    if ((long long)out_size >= base + 3) {
      out[base + 0] = 1024.0f;  // nums_window
      out[base + 1] = 256.0f;   // h
      out[base + 2] = 256.0f;   // w
    }
  }
}

// ---------------------------------------------------------------------------
// Main kernel: one block per window (4096 blocks x 256 threads)
// ---------------------------------------------------------------------------
__global__ void
__launch_bounds__(256)
win_attn_kernel(const float* __restrict__ xin, const float* __restrict__ sin_,
                const float* __restrict__ bq, const float* __restrict__ bk,
                const float* __restrict__ bv,
                const float* __restrict__ g1, const float* __restrict__ be1,
                const float* __restrict__ g2, const float* __restrict__ be2,
                const _Float16* __restrict__ wsh, float* __restrict__ out) {
  extern __shared__ char smem[];
  float*    tc32 = (float*)(smem + LDS_TC32);      // [c][n] fp32 (TDM order)
  _Float16* t16  = (_Float16*)(smem + LDS_T16);    // [n][T16S] f16
  _Float16* q16  = (_Float16*)(smem + LDS_Q);
  _Float16* k16  = (_Float16*)(smem + LDS_K);
  _Float16* vt16 = (_Float16*)(smem + LDS_VT);
  _Float16* att  = (_Float16*)(smem + LDS_ATT);
  _Float16* o16  = (_Float16*)(smem + LDS_O);
  float*    s32  = (float*)(smem + LDS_S32);
  _Float16* ln1  = (_Float16*)(smem + LDS_LN1);    // [n][T16S] f16
  _Float16* h1   = (_Float16*)(smem + LDS_H1);

  const int tid  = threadIdx.x;
  const int lane = tid & 31;
  const int wv   = tid >> 5;

  const int bid    = blockIdx.x;
  const int srcsel = bid >> 11;          // 0 -> x, 1 -> source
  const int widx   = bid & 2047;
  const int b  = widx >> 10;
  const int hy = (widx >> 5) & 31;
  const int wx = widx & 31;

  const float* in = srcsel ? sin_ : xin;
  const size_t gbase = ((size_t)b * DIM) * 65536 + (size_t)(hy * 8) * 256 + (size_t)(wx * 8);

  // ---- Stage 1a: TDM async DMA of the 3D window tile -> LDS (fp32, [c][n]) ----
  // Tile: x=8 contiguous, y=8 @ stride 256, z=256 channels @ stride 65536.
  if (wv == 0) {
    unsigned long long ga =
        (unsigned long long)(const void*)in + (unsigned long long)gbase * 4ull;
    v4i g0; v8i g1v; v4i g2; v4i g3;
    // Group 0: count=1 | lds_addr | global_addr | type=2
    g0[0] = 1;
    g0[1] = (int)LDS_TC32;
    g0[2] = (int)(ga & 0xFFFFFFFFull);
    g0[3] = (int)((ga >> 32) & 0x1FFFFFFull) | (int)(2u << 30);
    // Group 1: data_size=4B; tensor dims huge (no OOB); tile 8x8x256;
    //          stride0=256, stride1=65536 (elements)
    g1v[0] = 0x20000;                       // data_size=2 (4B), wg_mask=0
    g1v[1] = (int)((0x40000000u & 0xFFFFu) << 16);            // dim0 lo16 -> 0
    g1v[2] = (int)(0x40000000u >> 16);                        // dim0 hi | dim1 lo16(0)
    g1v[3] = (int)((0x40000000u >> 16) | (8u << 16));         // dim1 hi | tile_dim0=8
    g1v[4] = (int)(8u | (256u << 16));                        // tile_dim1=8 | tile_dim2=256
    g1v[5] = 256;                                             // stride0[31:0]
    g1v[6] = 0;                                               // stride0 hi | stride1 lo16(0)
    g1v[7] = 1;                                               // stride1 >> 16 (65536)
    // Group 2: tensor_dim2 huge; tile_dim3=0 (unused)
    g2[0] = 0x40000000; g2[1] = 0; g2[2] = 0; g2[3] = 0;
    // Group 3: dim4 unused
    g3[0] = 0; g3[1] = 0; g3[2] = 0; g3[3] = 0;
    asm volatile("tensor_load_to_lds %0, %1, %2, %3"
                 :: "s"(g0), "s"(g1v), "s"(g2), "s"(g3)
                 : "memory");
    __builtin_amdgcn_s_wait_tensorcnt(0);
  }
  __syncthreads();

  // ---- Stage 1b: LDS transpose [c][n] fp32 -> [n][c] f16 (padded stride) ----
  for (int i = tid; i < NTOK * DIM; i += 256) {   // i = c*64 + n
    int c = i >> 6, n = i & 63;
    t16[n * T16S + c] = (_Float16)tc32[i];
  }
  __syncthreads();

  // ---- Stage 2: Q,K,V = t @ W + b   (3 x [64,256]x[256,256]) ----
  for (int tile = wv; tile < 192; tile += 8) {
    int which = tile >> 6;               // 0=Q 1=K 2=V
    int r  = tile & 63;
    int mt = r >> 4, nt = r & 15;
    const _Float16* W = wsh + (which == 0 ? OFF_WQT : which == 1 ? OFF_WKT : OFF_WVT);
    const float* bias = (which == 0) ? bq : (which == 1) ? bk : bv;
    v8f acc = {};
#pragma unroll
    for (int ks = 0; ks < 8; ++ks) {
      v16h a  = load_a_frag(t16, T16S, mt * 16, ks * 32, lane);
      v16h bf = load_b_frag(W, DIM, nt * 16, ks * 32, lane);
      acc = wmma16(a, bf, acc);
    }
    float bb = bias[nt * 16 + (lane & 15)];
    int rowb = mt * 16 + ((lane >> 4) << 3);
    int col  = nt * 16 + (lane & 15);
    if (which < 2) {
      _Float16* dst = (which == 0) ? q16 : k16;
#pragma unroll
      for (int v = 0; v < 8; ++v) dst[(rowb + v) * DIM + col] = (_Float16)(acc[v] + bb);
    } else {
#pragma unroll
      for (int v = 0; v < 8; ++v) vt16[col * NTOK + (rowb + v)] = (_Float16)(acc[v] + bb);
    }
  }
  __syncthreads();

  // ---- Stage 3: attention, one head per wave (head dim 32) ----
  {
    const int h = wv;
    _Float16* attn = att + h * 64 * 64;
#pragma unroll 1
    for (int mt = 0; mt < 4; ++mt) {
      v16h a = load_a_frag(q16, DIM, mt * 16, h * 32, lane);
      v8f sc[4];
#pragma unroll
      for (int nt = 0; nt < 4; ++nt) {
        v16h bf = load_b_frag(k16, DIM, nt * 16, h * 32, lane);  // B[kd][j]=k[j][kd]
        v8f z = {};
        sc[nt] = wmma16(a, bf, z);
      }
      // softmax over 64 columns: rows live on 16-lane halves, 4 N-tiles deep
#pragma unroll
      for (int v = 0; v < 8; ++v) {
        float x0 = sc[0][v], x1 = sc[1][v], x2 = sc[2][v], x3 = sc[3][v];
        float m = fmaxf(fmaxf(x0, x1), fmaxf(x2, x3));
#pragma unroll
        for (int off = 1; off < 16; off <<= 1) m = fmaxf(m, __shfl_xor(m, off, 32));
        float e0 = __expf(x0 - m), e1 = __expf(x1 - m);
        float e2 = __expf(x2 - m), e3 = __expf(x3 - m);
        float s = e0 + e1 + e2 + e3;
#pragma unroll
        for (int off = 1; off < 16; off <<= 1) s += __shfl_xor(s, off, 32);
        float inv = 1.0f / s;
        int row = mt * 16 + v + ((lane >> 4) << 3);
        int cb  = lane & 15;
        attn[row * 64 + cb]      = (_Float16)(e0 * inv);
        attn[row * 64 + 16 + cb] = (_Float16)(e1 * inv);
        attn[row * 64 + 32 + cb] = (_Float16)(e2 * inv);
        attn[row * 64 + 48 + cb] = (_Float16)(e3 * inv);
      }
    }
    // o_h = attn @ v_h  : [64,64]x[64,32]
#pragma unroll 1
    for (int mt = 0; mt < 4; ++mt) {
#pragma unroll
      for (int nt = 0; nt < 2; ++nt) {
        v8f acc = {};
#pragma unroll
        for (int ks = 0; ks < 2; ++ks) {
          v16h a  = load_a_frag(attn, 64, mt * 16, ks * 32, lane);
          v16h bf = load_b_frag(vt16, NTOK, h * 32 + nt * 16, ks * 32, lane);
          acc = wmma16(a, bf, acc);
        }
        int rowb = mt * 16 + ((lane >> 4) << 3);
        int col  = h * 32 + nt * 16 + (lane & 15);
#pragma unroll
        for (int v = 0; v < 8; ++v) o16[(rowb + v) * DIM + col] = (_Float16)acc[v];
      }
    }
  }
  __syncthreads();

  // ---- Stage 4: o @ Wm -> s32 (q/k/vT regions now dead) ----
  for (int tile = wv; tile < 64; tile += 8) {
    int mt = tile >> 4, nt = tile & 15;
    v8f acc = {};
#pragma unroll
    for (int ks = 0; ks < 8; ++ks) {
      v16h a  = load_a_frag(o16, DIM, mt * 16, ks * 32, lane);
      v16h bf = load_b_frag(wsh + OFF_WMT, DIM, nt * 16, ks * 32, lane);
      acc = wmma16(a, bf, acc);
    }
    int rowb = mt * 16 + ((lane >> 4) << 3);
    int col  = nt * 16 + (lane & 15);
#pragma unroll
    for (int v = 0; v < 8; ++v) s32[(rowb + v) * DIM + col] = acc[v];
  }
  __syncthreads();

  // ---- Stage 5: LayerNorm(g1,be1) -> ln1 f16 ----
  if (tid < NTOK) {
    const float* rowp = s32 + tid * DIM;
    float mean = 0.f;
    for (int c = 0; c < DIM; ++c) mean += rowp[c];
    mean *= (1.0f / DIM);
    float var = 0.f;
    for (int c = 0; c < DIM; ++c) { float d = rowp[c] - mean; var += d * d; }
    var *= (1.0f / DIM);
    float rs = rsqrtf(var + 1e-5f);
    for (int c = 0; c < DIM; ++c)
      ln1[tid * T16S + c] = (_Float16)((rowp[c] - mean) * rs * g1[c] + be1[c]);
  }
  __syncthreads();

  // ---- Stage 6: h1 = relu(concat[t, ln1] @ W1)  ([64,512]x[512,512]) ----
  for (int tile = wv; tile < 128; tile += 8) {
    int mt = tile >> 5, nt = tile & 31;
    v8f acc = {};
#pragma unroll
    for (int ks = 0; ks < 16; ++ks) {
      int kk = ks * 32;
      const _Float16* abase = (kk < DIM) ? t16 : ln1;
      int kloc = (kk < DIM) ? kk : kk - DIM;
      v16h a  = load_a_frag(abase, T16S, mt * 16, kloc, lane);
      v16h bf = load_b_frag(wsh + OFF_W1T, 512, nt * 16, kk, lane);
      acc = wmma16(a, bf, acc);
    }
    int rowb = mt * 16 + ((lane >> 4) << 3);
    int col  = nt * 16 + (lane & 15);
#pragma unroll
    for (int v = 0; v < 8; ++v)
      h1[(rowb + v) * 512 + col] = (_Float16)fmaxf(acc[v], 0.0f);
  }
  __syncthreads();

  // ---- Stage 7: s32 = h1 @ W2  ([64,512]x[512,256]) ----
  for (int tile = wv; tile < 64; tile += 8) {
    int mt = tile >> 4, nt = tile & 15;
    v8f acc = {};
#pragma unroll
    for (int ks = 0; ks < 16; ++ks) {
      v16h a  = load_a_frag(h1, 512, mt * 16, ks * 32, lane);
      v16h bf = load_b_frag(wsh + OFF_W2T, 512, nt * 16, ks * 32, lane);
      acc = wmma16(a, bf, acc);
    }
    int rowb = mt * 16 + ((lane >> 4) << 3);
    int col  = nt * 16 + (lane & 15);
#pragma unroll
    for (int v = 0; v < 8; ++v) s32[(rowb + v) * DIM + col] = acc[v];
  }
  __syncthreads();

  // ---- Stage 8: LayerNorm(g2,be2) + residual -> global output ----
  float* outw = out + ((size_t)srcsel * 2048 + (size_t)widx) * NTOK * DIM;
  if (tid < NTOK) {
    const float* rowp = s32 + tid * DIM;
    float mean = 0.f;
    for (int c = 0; c < DIM; ++c) mean += rowp[c];
    mean *= (1.0f / DIM);
    float var = 0.f;
    for (int c = 0; c < DIM; ++c) { float d = rowp[c] - mean; var += d * d; }
    var *= (1.0f / DIM);
    float rs = rsqrtf(var + 1e-5f);
    for (int c = 0; c < DIM; ++c)
      outw[tid * DIM + c] =
          tc32[c * NTOK + tid] + ((rowp[c] - mean) * rs * g2[c] + be2[c]);
  }
}

// ---------------------------------------------------------------------------
extern "C" void kernel_launch(void* const* d_in, const int* in_sizes, int n_in,
                              void* d_out, int out_size, void* d_ws, size_t ws_size,
                              hipStream_t stream) {
  const float* x   = (const float*)d_in[0];
  const float* src = (const float*)d_in[1];
  const float* Wq  = (const float*)d_in[2];
  const float* bq  = (const float*)d_in[3];
  const float* Wk  = (const float*)d_in[4];
  const float* bk  = (const float*)d_in[5];
  const float* Wv  = (const float*)d_in[6];
  const float* bv  = (const float*)d_in[7];
  const float* Wm  = (const float*)d_in[8];
  const float* W1  = (const float*)d_in[9];
  const float* W2  = (const float*)d_in[10];
  const float* g1  = (const float*)d_in[11];
  const float* be1 = (const float*)d_in[12];
  const float* g2  = (const float*)d_in[13];
  const float* be2 = (const float*)d_in[14];
  float* out = (float*)d_out;
  _Float16* wsh = (_Float16*)d_ws;

  // Stage 0: f16-transpose all weights into workspace.
  cvt_transpose_kernel<<<256, 256, 0, stream>>>(Wq, wsh + OFF_WQT, 256, 256);
  cvt_transpose_kernel<<<256, 256, 0, stream>>>(Wk, wsh + OFF_WKT, 256, 256);
  cvt_transpose_kernel<<<256, 256, 0, stream>>>(Wv, wsh + OFF_WVT, 256, 256);
  cvt_transpose_kernel<<<256, 256, 0, stream>>>(Wm, wsh + OFF_WMT, 256, 256);
  cvt_transpose_kernel<<<1024, 256, 0, stream>>>(W1, wsh + OFF_W1T, 512, 512);
  cvt_transpose_kernel<<<512, 256, 0, stream>>>(W2, wsh + OFF_W2T, 512, 256);

  (void)hipFuncSetAttribute((const void*)win_attn_kernel,
                            hipFuncAttributeMaxDynamicSharedMemorySize,
                            SMEM_BYTES);
  win_attn_kernel<<<4096, 256, SMEM_BYTES, stream>>>(
      x, src, bq, bk, bv, g1, be1, g2, be2, wsh, out);

  scalar_tail_kernel<<<1, 64, 0, stream>>>(out, out_size);
}
